// AVWDCRNN_6640019440009
// MI455X (gfx1250) — matrix operand order repaired and checked by reading, not compiled, not run on hardware
//
#include <hip/hip_runtime.h>

// ---------------------------------------------------------------------------
// AVWDCRNN (2-layer adaptive graph GRU) for gfx1250 (MI455X), wave32 + WMMA.
// All GEMMs use v_wmma_f32_16x16x32_bf16 (bf16 operands, f32 accumulate).
// ---------------------------------------------------------------------------

typedef __bf16 bf16_t;
typedef __attribute__((ext_vector_type(16))) __bf16 v16bf;
typedef __attribute__((ext_vector_type(8)))  float  v8f;
typedef __attribute__((ext_vector_type(4)))  unsigned u32x4;

union BF16x16 { u32x4 q[2]; v16bf v; };
union BFCV { unsigned short s; bf16_t b; };

__device__ __forceinline__ bf16_t f2bf(float f) {
  unsigned u = __float_as_uint(f);
  unsigned r = (u + 0x7FFFu + ((u >> 16) & 1u)) >> 16;   // round-to-nearest-even
  BFCV cv; cv.s = (unsigned short)r; return cv.b;
}

__device__ __forceinline__ v8f wmma_bf(v16bf a, v16bf b, v8f c) {
  return __builtin_amdgcn_wmma_f32_16x16x32_bf16(
      /*neg_a=*/false, a, /*neg_b=*/false, b,
      /*c_mod=*/(short)0, c, /*reuse_a=*/false, /*reuse_b=*/false);
}

// A-operand (16x32 bf16, MxK): lane<16 holds row M=lane, K chunks {0..7,16..23};
// lane>=16 holds row M=lane-16, K chunks {8..15,24..31}. p -> element (M=0,K=0),
// ld = row stride in elements. All chunk addresses 16B aligned by construction.
__device__ __forceinline__ v16bf load_tileA(const bf16_t* p, int ld, int lane) {
  int l = lane & 15, hi = lane >> 4;
  const bf16_t* rp = p + (size_t)l * ld + hi * 8;
  BF16x16 u;
  u.q[0] = *(const u32x4*)(rp);
  u.q[1] = *(const u32x4*)(rp + 16);
  return u.v;
}

// B-operand (32x16 bf16, KxN): lane<16 holds col N=lane, K chunks {0..7,16..23};
// lane>=16 holds col N=lane-16, K chunks {8..15,24..31}. Storage is BT[col][k] (K contiguous),
// p -> element (K=0, col=0), ld = per-column stride in elements.
__device__ __forceinline__ v16bf load_tileB(const bf16_t* p, int ld, int lane) {
  int l = lane & 15, hi = lane >> 4;
  const bf16_t* rp = p + (size_t)l * ld + hi * 8;
  BF16x16 u;
  u.q[0] = *(const u32x4*)(rp);
  u.q[1] = *(const u32x4*)(rp + 16);
  return u.v;
}

// ---------------------------------------------------------------------------
// Setup kernels
// ---------------------------------------------------------------------------

// A = softmax(relu(E E^T), axis=1) -> bf16, row-major [n][m]. One block per row.
__global__ void k_supports(const float* __restrict__ E, bf16_t* __restrict__ A) {
  int n = blockIdx.x, tid = threadIdx.x;
  float e[16];
#pragma unroll
  for (int d = 0; d < 16; ++d) e[d] = E[n * 16 + d];
  float v[4];
#pragma unroll
  for (int j = 0; j < 4; ++j) {
    int m = tid + j * 256;
    float dot = 0.f;
#pragma unroll
    for (int d = 0; d < 16; ++d) dot += e[d] * E[m * 16 + d];
    v[j] = dot > 0.f ? dot : 0.f;
  }
  __shared__ float red[256];
  float mx = fmaxf(fmaxf(v[0], v[1]), fmaxf(v[2], v[3]));
  red[tid] = mx; __syncthreads();
  for (int s = 128; s > 0; s >>= 1) {
    if (tid < s) red[tid] = fmaxf(red[tid], red[tid + s]);
    __syncthreads();
  }
  mx = red[0]; __syncthreads();
  float sum = 0.f;
#pragma unroll
  for (int j = 0; j < 4; ++j) { v[j] = __expf(v[j] - mx); sum += v[j]; }
  red[tid] = sum; __syncthreads();
  for (int s = 128; s > 0; s >>= 1) {
    if (tid < s) red[tid] += red[tid + s];
    __syncthreads();
  }
  float inv = 1.f / red[0];
#pragma unroll
  for (int j = 0; j < 4; ++j)
    A[(size_t)n * 1024 + tid + j * 256] = f2bf(v[j] * inv);
}

// Wt[n][o][k][i] (bf16, K padded to Cpad per diffusion step, zero-filled pad rows)
//   = sum_d E[n,d] * gw[d][k][i][o]   for i < Creal, else 0.
__global__ void k_mkw(const float* __restrict__ E, const float* __restrict__ gw,
                      bf16_t* __restrict__ Wt, int Creal, int Cpad, int O, int total) {
  for (int idx = blockIdx.x * blockDim.x + threadIdx.x; idx < total;
       idx += gridDim.x * blockDim.x) {
    int i = idx % Cpad; int r1 = idx / Cpad;
    int k = r1 & 1;     int r2 = r1 >> 1;
    int o = r2 % O;     int n  = r2 / O;
    float val = 0.f;
    if (i < Creal) {
#pragma unroll
      for (int d = 0; d < 16; ++d)
        val += E[n * 16 + d] * gw[(((size_t)d * 2 + k) * Creal + i) * O + o];
    }
    Wt[idx] = f2bf(val);
  }
}

// bias[n][o] = sum_d E[n,d] * bp[d,o]
__global__ void k_mkb(const float* __restrict__ E, const float* __restrict__ bp,
                      float* __restrict__ bias, int O, int total) {
  for (int idx = blockIdx.x * blockDim.x + threadIdx.x; idx < total;
       idx += gridDim.x * blockDim.x) {
    int o = idx % O, n = idx / O;
    float v = 0.f;
#pragma unroll
    for (int d = 0; d < 16; ++d) v += E[n * 16 + d] * bp[d * O + o];
    bias[idx] = v;
  }
}

// state (B,N,HID) -> internal layout h[n][b][o]
__global__ void k_init(const float* __restrict__ init, float* __restrict__ h, int total) {
  for (int idx = blockIdx.x * blockDim.x + threadIdx.x; idx < total;
       idx += gridDim.x * blockDim.x) {
    int o = idx % 64; int nb = idx / 64;
    int b = nb % 64;  int n  = nb / 64;
    h[idx] = init[((size_t)b * 1024 + n) * 64 + o];
  }
}

// Build concat(x_t, state) into both GEMM layouts:
//   Xkm[b*Cpad + c][n]       (bf16; B-operand of diffusion GEMM, node-contiguous)
//   XG [n][b][k=0 slab c]    (bf16; A-operand of per-node GEMM, K-contiguous)
__global__ void k_build(const float* __restrict__ x, const float* __restrict__ xh,
                        const float* __restrict__ hstate,
                        bf16_t* __restrict__ Xkm, bf16_t* __restrict__ XG,
                        int t, int Cin, int Cpad, int Kp, int isl0, int total) {
  for (int idx = blockIdx.x * blockDim.x + threadIdx.x; idx < total;
       idx += gridDim.x * blockDim.x) {
    int c = idx % Cpad; int r = idx / Cpad;
    int b = r % 64;     int n = r / 64;
    float v = 0.f;
    if (c < Cin)
      v = isl0 ? x[(((size_t)b * 12 + t) * 1024 + n) * 2 + c]
               : xh[((size_t)n * 64 + b) * 64 + c];
    else if (c < Cin + 64)
      v = hstate[((size_t)n * 64 + b) * 64 + (c - Cin)];
    bf16_t bv = f2bf(v);
    Xkm[((size_t)(b * Cpad + c)) * 1024 + n]  = bv;
    XG[((size_t)n * 64 + b) * Kp + c]         = bv;
  }
}

// ---------------------------------------------------------------------------
// Diffusion GEMM: XG[n][b][Cpad + c] = sum_m A[n,m] * X[b*Cpad+c][m]  (bf16 out)
// Block = 256 thr (8 waves). Wave w: n-tile (blockIdx.y*8 + w), 4 col-tiles.
// c0 restricts recomputation to columns c >= c0 (tile-aligned).
// ---------------------------------------------------------------------------
__global__ void __launch_bounds__(256)
k_gemmA(const bf16_t* __restrict__ A, const bf16_t* __restrict__ X,
        bf16_t* XG, int Cpad, int c0, int Kp) {
  int lane = threadIdx.x & 31, wave = threadIdx.x >> 5;
  int W = Cpad - c0;
  int nbase = blockIdx.y * 128 + wave * 16;
  int jc0 = blockIdx.x * 64;
  int bb[4], cc[4], jrow[4];
#pragma unroll
  for (int jt = 0; jt < 4; ++jt) {
    int jc = jc0 + jt * 16;
    int b_ = jc / W;
    int c_ = c0 + (jc - b_ * W);
    bb[jt] = b_; cc[jt] = c_;
    jrow[jt] = b_ * Cpad + c_;
  }
  v8f acc[4] = {};
  for (int kk = 0; kk < 1024; kk += 32) {
    v16bf va = load_tileA(A + (size_t)nbase * 1024 + kk, 1024, lane);
#pragma unroll
    for (int jt = 0; jt < 4; ++jt) {
      v16bf vb = load_tileB(X + (size_t)jrow[jt] * 1024 + kk, 1024, lane);
      acc[jt] = wmma_bf(va, vb, acc[jt]);
    }
  }
  int col = lane & 15, rbase = (lane >> 4) * 8;
#pragma unroll
  for (int jt = 0; jt < 4; ++jt) {
#pragma unroll
    for (int r = 0; r < 8; ++r) {
      int nr = nbase + rbase + r;
      XG[((size_t)nr * 64 + bb[jt]) * Kp + Cpad + cc[jt] + col] = f2bf(acc[jt][r]);
    }
  }
}

// ---------------------------------------------------------------------------
// Per-node gate GEMM: ZR = sigmoid(XG[n] (64 x Kp) * Wt[n] (Kp x 128) + bg[n]).
// z (o<64): write z*state into state columns of Xkm and XG (k=0 slab).
// r (o>=64): stash into Rbuf. One block (8 waves) per node.
// ---------------------------------------------------------------------------
__global__ void __launch_bounds__(256)
k_gate(bf16_t* XG, const bf16_t* __restrict__ Wt,
       const float* __restrict__ bias, const float* __restrict__ hstate,
       float* __restrict__ Rbuf, bf16_t* Xkm,
       int Kp, int Cin, int Cpad) {
  int n = blockIdx.x;
  int lane = threadIdx.x & 31, wave = threadIdx.x >> 5;
  int mtile = wave & 3;
  int ob0 = (wave >> 2) * 64;
  const bf16_t* xg = XG + ((size_t)n * 64 + mtile * 16) * Kp;
  const bf16_t* wp = Wt + ((size_t)n * 128 + ob0) * Kp;
  v8f acc[4] = {};
  for (int kk = 0; kk < Kp; kk += 32) {
    v16bf va = load_tileA(xg + kk, Kp, lane);
#pragma unroll
    for (int ot = 0; ot < 4; ++ot) {
      v16bf vb = load_tileB(wp + (size_t)ot * 16 * Kp + kk, Kp, lane);
      acc[ot] = wmma_bf(va, vb, acc[ot]);
    }
  }
  __syncthreads();   // all waves done reading XG[n] before we overwrite its k=0 slab
  int col = lane & 15, rbase = (lane >> 4) * 8;
#pragma unroll
  for (int ot = 0; ot < 4; ++ot) {
    int o = ob0 + ot * 16 + col;
    float bv = bias[n * 128 + o];
#pragma unroll
    for (int r = 0; r < 8; ++r) {
      int brow = mtile * 16 + rbase + r;
      float s = 1.f / (1.f + __expf(-(acc[ot][r] + bv)));
      if (o < 64) {
        float zs = s * hstate[((size_t)n * 64 + brow) * 64 + o];
        bf16_t zb = f2bf(zs);
        Xkm[((size_t)(brow * Cpad + Cin + o)) * 1024 + n] = zb;
        XG[((size_t)n * 64 + brow) * Kp + Cin + o] = zb;
      } else {
        Rbuf[((size_t)n * 64 + brow) * 64 + (o - 64)] = s;
      }
    }
  }
}

// ---------------------------------------------------------------------------
// Per-node update GEMM: hc = tanh(XG[n]*Wu[n] + bu[n]); h = r*h + (1-r)*hc.
// Writes new state, optionally the output sequence and final-state slabs.
// ---------------------------------------------------------------------------
__global__ void __launch_bounds__(256)
k_update(const bf16_t* __restrict__ XG, const bf16_t* __restrict__ Wt,
         const float* __restrict__ bias, float* hstate,
         const float* __restrict__ Rbuf,
         float* outseq, float* outlast, int Kp, int t) {
  int n = blockIdx.x;
  int lane = threadIdx.x & 31, wave = threadIdx.x >> 5;
  int mtile = wave & 3;
  int ob0 = (wave >> 2) * 32;
  const bf16_t* xg = XG + ((size_t)n * 64 + mtile * 16) * Kp;
  const bf16_t* wp = Wt + ((size_t)n * 64 + ob0) * Kp;
  v8f acc[2] = {};
  for (int kk = 0; kk < Kp; kk += 32) {
    v16bf va = load_tileA(xg + kk, Kp, lane);
#pragma unroll
    for (int ot = 0; ot < 2; ++ot) {
      v16bf vb = load_tileB(wp + (size_t)ot * 16 * Kp + kk, Kp, lane);
      acc[ot] = wmma_bf(va, vb, acc[ot]);
    }
  }
  int col = lane & 15, rbase = (lane >> 4) * 8;
#pragma unroll
  for (int ot = 0; ot < 2; ++ot) {
    int o = ob0 + ot * 16 + col;
    float bv = bias[n * 64 + o];
#pragma unroll
    for (int r = 0; r < 8; ++r) {
      int brow = mtile * 16 + rbase + r;
      float hc = 1.f - 2.f / (1.f + __expf(2.f * (acc[ot][r] + bv)));  // tanh
      size_t si = ((size_t)n * 64 + brow) * 64 + o;
      float rg = Rbuf[si];
      float h = rg * hstate[si] + (1.f - rg) * hc;
      hstate[si] = h;
      if (outseq)  outseq[(((size_t)brow * 12 + t) * 1024 + n) * 64 + o] = h;
      if (outlast) outlast[((size_t)brow * 1024 + n) * 64 + o] = h;
    }
  }
}

// ---------------------------------------------------------------------------
// Host
// ---------------------------------------------------------------------------
extern "C" void kernel_launch(void* const* d_in, const int* in_sizes, int n_in,
                              void* d_out, int out_size, void* d_ws, size_t ws_size,
                              hipStream_t stream) {
  const float* x    = (const float*)d_in[0];   // (64,12,1024,2)
  const float* init = (const float*)d_in[1];   // (2,64,1024,64)
  const float* E    = (const float*)d_in[2];   // (1024,16)
  const float* gw0  = (const float*)d_in[3];   // (16,2,66,128)
  const float* gb0  = (const float*)d_in[4];   // (16,128)
  const float* uw0  = (const float*)d_in[5];   // (16,2,66,64)
  const float* ub0  = (const float*)d_in[6];   // (16,64)
  const float* gw1  = (const float*)d_in[7];   // (16,2,128,128)
  const float* gb1  = (const float*)d_in[8];   // (16,128)
  const float* uw1  = (const float*)d_in[9];   // (16,2,128,64)
  const float* ub1  = (const float*)d_in[10];  // (16,64)
  float* out = (float*)d_out;

  char* w = (char*)d_ws;
  size_t off = 0;
  auto alloc = [&](size_t bytes) {
    void* p = w + off;
    off = (off + bytes + 255) & ~(size_t)255;
    return p;
  };
  bf16_t* Abf  = (bf16_t*)alloc((size_t)1024 * 1024 * 2);
  bf16_t* Wg0  = (bf16_t*)alloc((size_t)1024 * 128 * 160 * 2);
  bf16_t* Wu0  = (bf16_t*)alloc((size_t)1024 * 64  * 160 * 2);
  bf16_t* Wg1  = (bf16_t*)alloc((size_t)1024 * 128 * 256 * 2);
  bf16_t* Wu1  = (bf16_t*)alloc((size_t)1024 * 64  * 256 * 2);
  float*  bg0  = (float*) alloc((size_t)1024 * 128 * 4);
  float*  bu0  = (float*) alloc((size_t)1024 * 64  * 4);
  float*  bg1  = (float*) alloc((size_t)1024 * 128 * 4);
  float*  bu1  = (float*) alloc((size_t)1024 * 64  * 4);
  bf16_t* Xkm  = (bf16_t*)alloc((size_t)8192 * 1024 * 2);
  bf16_t* XG   = (bf16_t*)alloc((size_t)1024 * 64 * 256 * 2);
  float*  Rbuf = (float*) alloc((size_t)1024 * 64 * 64 * 4);
  float*  h0   = (float*) alloc((size_t)1024 * 64 * 64 * 4);
  float*  h1   = (float*) alloc((size_t)1024 * 64 * 64 * 4);
  (void)ws_size; (void)in_sizes; (void)n_in; (void)out_size;

  // --- one-time setup ---
  k_supports<<<1024, 256, 0, stream>>>(E, Abf);
  k_mkw<<<4096, 256, 0, stream>>>(E, gw0, Wg0,  66,  80, 128, 1024 * 128 * 160);
  k_mkw<<<4096, 256, 0, stream>>>(E, uw0, Wu0,  66,  80,  64, 1024 * 64  * 160);
  k_mkw<<<4096, 256, 0, stream>>>(E, gw1, Wg1, 128, 128, 128, 1024 * 128 * 256);
  k_mkw<<<4096, 256, 0, stream>>>(E, uw1, Wu1, 128, 128,  64, 1024 * 64  * 256);
  k_mkb<<<256, 256, 0, stream>>>(E, gb0, bg0, 128, 1024 * 128);
  k_mkb<<<256, 256, 0, stream>>>(E, ub0, bu0,  64, 1024 * 64);
  k_mkb<<<256, 256, 0, stream>>>(E, gb1, bg1, 128, 1024 * 128);
  k_mkb<<<256, 256, 0, stream>>>(E, ub1, bu1,  64, 1024 * 64);
  k_init<<<2048, 256, 0, stream>>>(init,                     h0, 1024 * 64 * 64);
  k_init<<<2048, 256, 0, stream>>>(init + (size_t)64 * 1024 * 64, h1, 1024 * 64 * 64);

  float* cur   = out;                               // (B,T,N,HID)
  float* last0 = out + (size_t)64 * 12 * 1024 * 64; // lasts[0]
  float* last1 = last0 + (size_t)64 * 1024 * 64;    // lasts[1]

  dim3 gA0(80, 8), gA1(128, 8), gA1b(64, 8);
  for (int t = 0; t < 12; ++t) {
    // ----- layer 0: Cin=2, C=66 -> Cpad=80, Kp=160 -----
    k_build<<<2048, 256, 0, stream>>>(x, nullptr, h0, Xkm, XG, t, 2, 80, 160, 1,
                                      1024 * 64 * 80);
    k_gemmA<<<gA0, 256, 0, stream>>>(Abf, Xkm, XG, 80, 0, 160);
    k_gate<<<1024, 256, 0, stream>>>(XG, Wg0, bg0, h0, Rbuf, Xkm, 160, 2, 80);
    k_gemmA<<<gA0, 256, 0, stream>>>(Abf, Xkm, XG, 80, 0, 160);
    k_update<<<1024, 256, 0, stream>>>(XG, Wu0, bu0, h0, Rbuf, nullptr,
                                       (t == 11) ? last0 : nullptr, 160, t);
    // ----- layer 1: Cin=64, C=128 -> Cpad=128, Kp=256 -----
    k_build<<<2048, 256, 0, stream>>>(nullptr, h0, h1, Xkm, XG, t, 64, 128, 256, 0,
                                      1024 * 64 * 128);
    k_gemmA<<<gA1, 256, 0, stream>>>(Abf, Xkm, XG, 128, 0, 256);
    k_gate<<<1024, 256, 0, stream>>>(XG, Wg1, bg1, h1, Rbuf, Xkm, 256, 64, 128);
    // only state columns (c >= 64) changed -> recompute just those tiles
    k_gemmA<<<gA1b, 256, 0, stream>>>(Abf, Xkm, XG, 128, 64, 256);
    k_update<<<1024, 256, 0, stream>>>(XG, Wu1, bu1, h1, Rbuf, cur,
                                       (t == 11) ? last1 : nullptr, 256, t);
  }
}
